// NanoGPTAttention_18880676233455
// MI455X (gfx1250) — compile-verified, matrix operand order checked by reading, not run on hardware
//
#include <hip/hip_runtime.h>
#include <hip/hip_bf16.h>

// ---------------------------------------------------------------------------
// NanoGPT attention block for gfx1250 (MI455X):
//   qkv = x @ w_qkv + b_qkv ; flash causal attention ; out = attn @ w_out + b_out
// Strategy: bf16 WMMA (v_wmma_f32_16x16x32_bf16) everywhere, f32 accumulate.
// B=2, T=2048, C=1024, H=16, D=64.
// ---------------------------------------------------------------------------

typedef __bf16 v16bf __attribute__((ext_vector_type(16)));
typedef __bf16 v8bf  __attribute__((ext_vector_type(8)));
typedef float  v8f   __attribute__((ext_vector_type(8)));

#define BB 2
#define TT 2048
#define CC 1024
#define HH 16
#define DD 64
#define MM (BB * TT)   /* 4096 */
#define N3 (3 * CC)    /* 3072 */

// Load a 16x32 bf16 A/B fragment for one lane per CDNA5 WMMA layout:
// lane L: row = L&15, kbase = (L>>4)*8 ; elements 0..7 -> K=kbase..kbase+7,
// elements 8..15 -> K=kbase+16..kbase+23. Caller passes p = row_ptr + kbase.
__device__ __forceinline__ v16bf load_frag(const __bf16* p) {
    v8bf lo = *reinterpret_cast<const v8bf*>(p);
    v8bf hi = *reinterpret_cast<const v8bf*>(p + 16);
    return __builtin_shufflevector(lo, hi, 0, 1, 2, 3, 4, 5, 6, 7,
                                   8, 9, 10, 11, 12, 13, 14, 15);
}

__device__ __forceinline__ v8f wmma_bf16(v16bf a, v16bf b, v8f c) {
    return __builtin_amdgcn_wmma_f32_16x16x32_bf16(
        /*neg_a=*/false, a, /*neg_b=*/false, b,
        /*c_mod=*/(short)0, c, /*reuse_a=*/false, /*reuse_b=*/false);
}

__device__ __forceinline__ v8f zero_v8f() {
    v8f z = {0.f, 0.f, 0.f, 0.f, 0.f, 0.f, 0.f, 0.f};
    return z;
}

// ---------------------------------------------------------------------------
// Elementwise f32 -> bf16 convert.
// ---------------------------------------------------------------------------
__global__ void cvt_bf16_kernel(const float* __restrict__ in,
                                __bf16* __restrict__ out, int n) {
    int i = blockIdx.x * blockDim.x + threadIdx.x;
    int stride = gridDim.x * blockDim.x;
    for (; i < n; i += stride) out[i] = (__bf16)in[i];
}

// Transposed convert: in f32 [K][N] row-major -> out bf16 [N][K] row-major.
// Puts weights in the [N][K] layout so B fragments are contiguous 16B loads.
__global__ void transpose_cvt_kernel(const float* __restrict__ in,
                                     __bf16* __restrict__ out, int K, int N) {
    int i = blockIdx.x * blockDim.x + threadIdx.x;
    int stride = gridDim.x * blockDim.x;
    const int total = K * N;
    for (; i < total; i += stride) {
        const int k = i / N;
        const int n = i - k * N;
        out[(size_t)n * K + k] = (__bf16)in[i];
    }
}

// ---------------------------------------------------------------------------
// QKV GEMM: [4096,1024]bf16 @ [1024,3072] (Bt stored [3072,1024]) + bias,
// scattered into q/k/v buffers laid out [B,H,T,D] in bf16.
// 4 waves / WG, each wave owns a 64x64 accumulator tile (16 wmma / k-step).
// ---------------------------------------------------------------------------
__global__ __launch_bounds__(128) void qkv_gemm_kernel(
    const __bf16* __restrict__ A,    // [MM, CC]
    const __bf16* __restrict__ Bt,   // [N3, CC]
    const float* __restrict__ bias,  // [N3]
    __bf16* __restrict__ qb, __bf16* __restrict__ kb, __bf16* __restrict__ vb) {
    const int lane = threadIdx.x & 31;
    const int wave = threadIdx.x >> 5;
    const int lo = lane & 15, hi = lane >> 4;
    const int m0 = blockIdx.y * 128 + (wave >> 1) * 64;
    const int n0 = blockIdx.x * 128 + (wave & 1) * 64;

    v8f acc[4][4];
#pragma unroll
    for (int i = 0; i < 4; ++i)
#pragma unroll
        for (int j = 0; j < 4; ++j) acc[i][j] = zero_v8f();

    for (int k0 = 0; k0 < CC; k0 += 32) {
        const int kbase = k0 + hi * 8;
        if (k0 + 32 < CC) {  // global_prefetch_b8 the next k-tile
            __builtin_prefetch(A + (size_t)(m0 + lo) * CC + k0 + 32, 0, 1);
            __builtin_prefetch(Bt + (size_t)(n0 + lo) * CC + k0 + 32, 0, 1);
        }
        v16bf af[4], bf[4];
#pragma unroll
        for (int mt = 0; mt < 4; ++mt)
            af[mt] = load_frag(A + (size_t)(m0 + mt * 16 + lo) * CC + kbase);
#pragma unroll
        for (int nt = 0; nt < 4; ++nt)
            bf[nt] = load_frag(Bt + (size_t)(n0 + nt * 16 + lo) * CC + kbase);
#pragma unroll
        for (int mt = 0; mt < 4; ++mt)
#pragma unroll
            for (int nt = 0; nt < 4; ++nt)
                acc[mt][nt] = wmma_bf16(af[mt], bf[nt], acc[mt][nt]);
    }

    // Epilogue: C/D layout lane (lo,hi): element r -> row r+8*hi, col lo.
#pragma unroll
    for (int nt = 0; nt < 4; ++nt) {
        const int n = n0 + nt * 16 + lo;
        const int which = n >> 10;           // 0=q 1=k 2=v
        const int h = (n & (CC - 1)) >> 6;   // head
        const int d = n & (DD - 1);
        __bf16* dst = (which == 0) ? qb : ((which == 1) ? kb : vb);
        const float bv = bias[n];
#pragma unroll
        for (int mt = 0; mt < 4; ++mt)
#pragma unroll
            for (int r = 0; r < 8; ++r) {
                const int m = m0 + mt * 16 + r + 8 * hi;
                const int bidx = m >> 11;    // / TT
                const int t = m & (TT - 1);
                dst[(((size_t)(bidx * HH + h) * TT + t) << 6) + d] =
                    (__bf16)(acc[mt][nt][r] + bv);
            }
    }
}

// ---------------------------------------------------------------------------
// Flash causal attention. Grid: (T/64 q-blocks, B*H). 4 waves/WG, each wave
// owns a 16-row query strip. Scores stay in WMMA accumulators; online
// softmax; P staged through LDS to reach the A-operand layout; V transposed
// into LDS so its B fragments are contiguous.
// ---------------------------------------------------------------------------
__global__ __launch_bounds__(128) void flash_attn_kernel(
    const __bf16* __restrict__ qb, const __bf16* __restrict__ kb,
    const __bf16* __restrict__ vb, __bf16* __restrict__ attn) {
    __shared__ __align__(16) __bf16 Vt[64][64];        // Vt[d][j]
    __shared__ __align__(16) __bf16 Pst[4][16][64];    // per-wave P staging

    const int lane = threadIdx.x & 31;
    const int wave = threadIdx.x >> 5;
    const int lo = lane & 15, hi = lane >> 4;
    const int qblk = blockIdx.x;
    const int bh = blockIdx.y;
    const int b = bh >> 4, h = bh & 15;
    const int q0 = qblk * 64;
    const int r0 = q0 + wave * 16;   // this wave's query-strip start row

    const size_t base = (size_t)bh * TT * DD;
    const __bf16* qp = qb + base;
    const __bf16* kp = kb + base;
    const __bf16* vp = vb + base;

    // Q strip [16 x 64] as two 16x32 A fragments, kept in registers.
    v16bf qa[2];
#pragma unroll
    for (int ks = 0; ks < 2; ++ks)
        qa[ks] = load_frag(qp + (size_t)(r0 + lo) * DD + hi * 8 + ks * 32);

    v8f acc[4];
#pragma unroll
    for (int i = 0; i < 4; ++i) acc[i] = zero_v8f();
    float m_i[8], l_i[8];
#pragma unroll
    for (int r = 0; r < 8; ++r) { m_i[r] = -1e30f; l_i[r] = 0.f; }

    for (int jb = 0; jb <= q0; jb += 64) {
        __syncthreads();  // previous iteration's Vt reads complete
        {   // cooperative transposed load of V block into LDS
            const int jr = threadIdx.x >> 1;        // 0..63
            const int d0 = (threadIdx.x & 1) * 32;  // 0 or 32
            const __bf16* src = vp + (size_t)(jb + jr) * DD + d0;
#pragma unroll
            for (int c = 0; c < 32; ++c) Vt[d0 + c][jr] = src[c];
        }
        __syncthreads();

        // S strip [16 x 64] = Q @ K^T ; K rows are already the [N][K] layout.
        v8f s[4];
#pragma unroll
        for (int i = 0; i < 4; ++i) s[i] = zero_v8f();
#pragma unroll
        for (int ks = 0; ks < 2; ++ks) {
#pragma unroll
            for (int nt = 0; nt < 4; ++nt) {
                v16bf kf = load_frag(
                    kp + (size_t)(jb + nt * 16 + lo) * DD + hi * 8 + ks * 32);
                s[nt] = wmma_bf16(qa[ks], kf, s[nt]);
            }
        }

        // scale (1/sqrt(64)) + causal mask
#pragma unroll
        for (int nt = 0; nt < 4; ++nt)
#pragma unroll
            for (int r = 0; r < 8; ++r) {
                const int row = r0 + r + 8 * hi;
                const int col = jb + nt * 16 + lo;
                const float v = s[nt][r] * 0.125f;
                s[nt][r] = (col > row) ? -1e30f : v;
            }

        // online softmax: rows live across lanes 0-15 (r) / 16-31 (r+8),
        // so xor-shuffles 1,2,4,8 reduce within each half independently.
#pragma unroll
        for (int r = 0; r < 8; ++r) {
            float mx = s[0][r];
#pragma unroll
            for (int nt = 1; nt < 4; ++nt) mx = fmaxf(mx, s[nt][r]);
#pragma unroll
            for (int off = 1; off <= 8; off <<= 1)
                mx = fmaxf(mx, __shfl_xor(mx, off, 32));
            const float mnew = fmaxf(m_i[r], mx);
            const float alpha = __expf(m_i[r] - mnew);
            m_i[r] = mnew;
            float ps = 0.f;
#pragma unroll
            for (int nt = 0; nt < 4; ++nt) {
                const float p = __expf(s[nt][r] - mnew);
                s[nt][r] = p;
                ps += p;
            }
#pragma unroll
            for (int off = 1; off <= 8; off <<= 1)
                ps += __shfl_xor(ps, off, 32);
            l_i[r] = l_i[r] * alpha + ps;
#pragma unroll
            for (int dt = 0; dt < 4; ++dt) acc[dt][r] *= alpha;
        }

        // stage P (bf16) through LDS to reach the A-operand layout
#pragma unroll
        for (int nt = 0; nt < 4; ++nt)
#pragma unroll
            for (int r = 0; r < 8; ++r)
                Pst[wave][r + 8 * hi][nt * 16 + lo] = (__bf16)s[nt][r];

        // O += P @ V  (A = staged P, B = transposed V in LDS)
#pragma unroll
        for (int ks = 0; ks < 2; ++ks) {
            v16bf pf = load_frag(&Pst[wave][lo][hi * 8 + ks * 32]);
#pragma unroll
            for (int dt = 0; dt < 4; ++dt) {
                v16bf vf = load_frag(&Vt[dt * 16 + lo][hi * 8 + ks * 32]);
                acc[dt] = wmma_bf16(pf, vf, acc[dt]);
            }
        }
    }

    // finalize: O /= l, write attn_out [B,T,C] bf16
#pragma unroll
    for (int r = 0; r < 8; ++r) l_i[r] = 1.f / l_i[r];
#pragma unroll
    for (int dt = 0; dt < 4; ++dt)
#pragma unroll
        for (int r = 0; r < 8; ++r) {
            const int t = r0 + r + 8 * hi;
            const int c = h * DD + dt * 16 + lo;
            attn[(size_t)(b * TT + t) * CC + c] = (__bf16)(acc[dt][r] * l_i[r]);
        }
}

// ---------------------------------------------------------------------------
// Output projection: attn[4096,1024]bf16 @ w_out (Bt [1024,1024]) + b_out,
// f32 result into d_out.
// ---------------------------------------------------------------------------
__global__ __launch_bounds__(128) void out_gemm_kernel(
    const __bf16* __restrict__ A,    // [MM, CC]
    const __bf16* __restrict__ Bt,   // [CC, CC]
    const float* __restrict__ bias,  // [CC]
    float* __restrict__ out) {       // [MM, CC]
    const int lane = threadIdx.x & 31;
    const int wave = threadIdx.x >> 5;
    const int lo = lane & 15, hi = lane >> 4;
    const int m0 = blockIdx.y * 128 + (wave >> 1) * 64;
    const int n0 = blockIdx.x * 128 + (wave & 1) * 64;

    v8f acc[4][4];
#pragma unroll
    for (int i = 0; i < 4; ++i)
#pragma unroll
        for (int j = 0; j < 4; ++j) acc[i][j] = zero_v8f();

    for (int k0 = 0; k0 < CC; k0 += 32) {
        const int kbase = k0 + hi * 8;
        if (k0 + 32 < CC) {
            __builtin_prefetch(A + (size_t)(m0 + lo) * CC + k0 + 32, 0, 1);
            __builtin_prefetch(Bt + (size_t)(n0 + lo) * CC + k0 + 32, 0, 1);
        }
        v16bf af[4], bf[4];
#pragma unroll
        for (int mt = 0; mt < 4; ++mt)
            af[mt] = load_frag(A + (size_t)(m0 + mt * 16 + lo) * CC + kbase);
#pragma unroll
        for (int nt = 0; nt < 4; ++nt)
            bf[nt] = load_frag(Bt + (size_t)(n0 + nt * 16 + lo) * CC + kbase);
#pragma unroll
        for (int mt = 0; mt < 4; ++mt)
#pragma unroll
            for (int nt = 0; nt < 4; ++nt)
                acc[mt][nt] = wmma_bf16(af[mt], bf[nt], acc[mt][nt]);
    }

#pragma unroll
    for (int nt = 0; nt < 4; ++nt) {
        const int n = n0 + nt * 16 + lo;
        const float bv = bias[n];
#pragma unroll
        for (int mt = 0; mt < 4; ++mt)
#pragma unroll
            for (int r = 0; r < 8; ++r) {
                const int m = m0 + mt * 16 + r + 8 * hi;
                out[(size_t)m * CC + n] = acc[mt][nt][r] + bv;
            }
    }
}

// ---------------------------------------------------------------------------
// Host-side launch wiring. Workspace layout (bytes):
//   [0,        8388608)  x_bf16   [4096,1024]
//   [8388608, 14680064)  wqkv_t   [3072,1024]
//   [14680064,16777216)  wout_t   [1024,1024]
//   [16777216,25165824)  q        [B,H,T,D]
//   [25165824,33554432)  k        [B,H,T,D]
//   [33554432,41943040)  v        [B,H,T,D]
//   [41943040,50331648)  attn     [4096,1024]
// ---------------------------------------------------------------------------
extern "C" void kernel_launch(void* const* d_in, const int* in_sizes, int n_in,
                              void* d_out, int out_size, void* d_ws,
                              size_t ws_size, hipStream_t stream) {
    const float* x     = (const float*)d_in[0];
    const float* w_qkv = (const float*)d_in[1];
    const float* b_qkv = (const float*)d_in[2];
    const float* w_out = (const float*)d_in[3];
    const float* b_out = (const float*)d_in[4];
    float* out = (float*)d_out;

    char* ws = (char*)d_ws;
    __bf16* x_bf   = (__bf16*)(ws);
    __bf16* wqkv_t = (__bf16*)(ws + 8388608);
    __bf16* wout_t = (__bf16*)(ws + 14680064);
    __bf16* qbuf   = (__bf16*)(ws + 16777216);
    __bf16* kbuf   = (__bf16*)(ws + 25165824);
    __bf16* vbuf   = (__bf16*)(ws + 33554432);
    __bf16* attn   = (__bf16*)(ws + 41943040);

    cvt_bf16_kernel<<<2048, 256, 0, stream>>>(x, x_bf, MM * CC);
    transpose_cvt_kernel<<<2048, 256, 0, stream>>>(w_qkv, wqkv_t, CC, N3);
    transpose_cvt_kernel<<<1024, 256, 0, stream>>>(w_out, wout_t, CC, CC);

    qkv_gemm_kernel<<<dim3(N3 / 128, MM / 128), 128, 0, stream>>>(
        x_bf, wqkv_t, b_qkv, qbuf, kbuf, vbuf);
    flash_attn_kernel<<<dim3(TT / 64, BB * HH), 128, 0, stream>>>(
        qbuf, kbuf, vbuf, attn);
    out_gemm_kernel<<<dim3(CC / 128, MM / 128), 128, 0, stream>>>(
        attn, wout_t, b_out, out);
}
